// TransformerLayer_2619930051047
// MI455X (gfx1250) — matrix operand order repaired and checked by reading, not compiled, not run on hardware
//
#include <hip/hip_runtime.h>
#include <math.h>

// ---------------------------------------------------------------------------
// Types for CDNA5 WMMA (wave32): bf16 A/B fragments (v16bf) and f32 C/D (v8f).
// ---------------------------------------------------------------------------
typedef __bf16 bf16;
typedef __bf16 v16bf __attribute__((ext_vector_type(16)));
typedef __bf16 v8bf  __attribute__((ext_vector_type(8)));
typedef float  v8f   __attribute__((ext_vector_type(8)));

// Problem constants (from setup_inputs): b=2, h=w=128, d=256, ns=4, f=4
#define DIM    256
#define NWIN   32          // b * ns * ns
#define MROWS  32768       // b*h*w
#define DHID   2048        // 2*d*f
#define DCAT   512         // 2*d

// ---------------------------------------------------------------------------
// Window shift+split index map. roll(-16) before split and roll(+16) after
// merge produce the SAME map, so this serves both gather (QKV) and scatter
// (merge un-window).
// ---------------------------------------------------------------------------
__device__ __forceinline__ int win_map(int r) {
  int win = r >> 10, n = r & 1023;
  int bb = win >> 4, i = (win >> 2) & 3, j = win & 3;
  int y = (n >> 5) & 31, x = n & 31;
  int h = ((i << 5) + y + 16) & 127;
  int w = ((j << 5) + x + 16) & 127;
  return (bb << 14) + (h << 7) + w;
}

// ---------------------------------------------------------------------------
// f32 -> bf16 conversion, optional output column offset/stride (for hcat).
// ---------------------------------------------------------------------------
__global__ void cvt_strided(const float* __restrict__ in, bf16* __restrict__ out,
                            int rows, int cols, int ldo, int off) {
  int i = blockIdx.x * 256 + threadIdx.x;
  if (i >= rows * cols) return;
  int r = i / cols, c = i - r * cols;
  out[(long)r * ldo + off + c] = (bf16)in[i];
}

// f32 [K,N] row-major -> bf16 [N,K] row-major (weight pre-transpose so that
// WMMA B-fragments (fixed n per lane, contiguous k) are 16B vector loads).
__global__ void cvt_transpose(const float* __restrict__ in, bf16* __restrict__ out,
                              int K, int N) {
  int i = blockIdx.x * 256 + threadIdx.x;
  if (i >= K * N) return;
  int n = i / K, k = i - n * K;
  out[i] = (bf16)in[(long)k * N + n];
}

// ---------------------------------------------------------------------------
// Generic bf16 WMMA GEMM:  C[M,N] = act( alpha * A[M,K] @ Bt[N,K]^T )
//   block = 256 threads (8 waves), workgroup tile 128x128, wave tile 32x64,
//   K-step 32, f32 accumulation via v_wmma_f32_16x16x32_bf16.
// GATHER : 1 -> A rows gathered through win_map (QKV with shift+window)
// SMODE  : 0 bf16 row-major | 1 f32 row-major | 2 f32 rows scattered via
//          win_map (merge un-window) | 3 bf16 window-transposed
//          [win][feat(256)][key(1024)] (produces V^T for P@V B-frags)
// ACT    : 1 -> exact GELU epilogue
// ---------------------------------------------------------------------------
template <int GATHER, int SMODE, int ACT>
__launch_bounds__(256)
__global__ void gemm_wmma(const bf16* __restrict__ A, int lda,
                          const bf16* __restrict__ Bt, int K,
                          void* __restrict__ Cout, int ldc, float alpha) {
  const int lane = threadIdx.x & 31;
  const int wave = threadIdx.x >> 5;
  const int lrow = lane & 15;
  const int lhi  = lane >> 4;
  const int M0 = blockIdx.x * 128 + (wave >> 1) * 32;
  const int N0 = blockIdx.y * 128 + (wave & 1) * 64;

  const v8f zero8 = {0.f, 0.f, 0.f, 0.f, 0.f, 0.f, 0.f, 0.f};
  v8f acc[2][4];
#pragma unroll
  for (int i = 0; i < 2; ++i)
#pragma unroll
    for (int j = 0; j < 4; ++j) acc[i][j] = zero8;

  const bf16* aptr[2];
#pragma unroll
  for (int i = 0; i < 2; ++i) {
    int r = M0 + i * 16 + lrow;
    int g = GATHER ? win_map(r) : r;
    aptr[i] = A + (long)g * lda;
  }
  const bf16* bptr[4];
#pragma unroll
  for (int j = 0; j < 4; ++j) bptr[j] = Bt + (long)(N0 + j * 16 + lrow) * K;

  for (int kk = 0; kk < K; kk += 32) {
    v16bf a[2], b[4];
#pragma unroll
    for (int i = 0; i < 2; ++i) {
      const bf16* p = aptr[i] + kk + lhi * 8;   // A frag: m=lane%16, k table order
      __builtin_prefetch(p + 64, 0, 1);         // global_prefetch_b8
      v8bf lo = *(const v8bf*)p;
      v8bf hi = *(const v8bf*)(p + 16);
#pragma unroll
      for (int e = 0; e < 8; ++e) { a[i][e] = lo[e]; a[i][e + 8] = hi[e]; }
    }
#pragma unroll
    for (int j = 0; j < 4; ++j) {
      const bf16* p = bptr[j] + kk + lhi * 8;   // B frag: n=lane%16, k table order
      v8bf lo = *(const v8bf*)p;
      v8bf hi = *(const v8bf*)(p + 16);
#pragma unroll
      for (int e = 0; e < 8; ++e) { b[j][e] = lo[e]; b[j][e + 8] = hi[e]; }
    }
#pragma unroll
    for (int i = 0; i < 2; ++i)
#pragma unroll
      for (int j = 0; j < 4; ++j)
        acc[i][j] = __builtin_amdgcn_wmma_f32_16x16x32_bf16(
            false, a[i], false, b[j], (short)0, acc[i][j], false, false);
  }

#pragma unroll
  for (int i = 0; i < 2; ++i)
#pragma unroll
    for (int j = 0; j < 4; ++j)
#pragma unroll
      for (int q = 0; q < 8; ++q) {
        int m = M0 + i * 16 + lhi * 8 + q;      // C layout: lane-half picks rows
        int n = N0 + j * 16 + lrow;
        float v = acc[i][j][q] * alpha;
        if (ACT) v = 0.5f * v * (1.0f + erff(v * 0.70710678118f));
        if (SMODE == 0) {
          ((bf16*)Cout)[(long)m * ldc + n] = (bf16)v;
        } else if (SMODE == 1) {
          ((float*)Cout)[(long)m * ldc + n] = v;
        } else if (SMODE == 2) {
          ((float*)Cout)[(long)win_map(m) * ldc + n] = v;
        } else {  // V^T: [win][feat][key]
          int win = m >> 10, key = m & 1023;
          ((bf16*)Cout)[((long)win << 18) + ((long)n << 10) + key] = (bf16)v;
        }
      }
}

// ---------------------------------------------------------------------------
// Flash-style shifted-window attention with async Global->LDS K/V staging.
//   grid = 32 windows * 8 row-slabs; block = 256 threads (8 waves).
//   Per 32-key step, the workgroup cooperatively stages the 16KB K tile and
//   16KB V^T tile into double-buffered LDS with GLOBAL_LOAD_ASYNC_TO_LDS_B128
//   (ASYNCcnt), overlapping the next tile's DMA with current-tile WMMAs.
//   LDS B-fragments are batch-loaded (clauses of ds_load_b128) ahead of each
//   WMMA burst so a single DScnt wait covers the batch.
// ---------------------------------------------------------------------------
__launch_bounds__(256)
__global__ void swin_attn(const bf16* __restrict__ qw, const bf16* __restrict__ kw,
                          const bf16* __restrict__ vt, const float* __restrict__ mask,
                          bf16* __restrict__ out) {
  __shared__ bf16 kbuf[2][32 * 256];   // [key][feat]  16KB per buffer
  __shared__ bf16 vbuf[2][256 * 32];   // [feat][key]  16KB per buffer
  __shared__ bf16 plds[8][16 * 32];    // per-wave P staging (C->A relayout)

  const int tid  = threadIdx.x;
  const int lane = tid & 31;
  const int wave = tid >> 5;
  const int lrow = lane & 15;
  const int lhi  = lane >> 4;
  const int win  = blockIdx.x >> 3;
  const int rowInWin = (blockIdx.x & 7) * 128 + wave * 16;

  const bf16* qbase = qw + (long)(win * 1024 + rowInWin + lrow) * 256;
  const bf16* kbase = kw + (long)win * 1024 * 256;
  const bf16* vbase = vt + ((long)win << 18);
  const float* mbase = mask + (long)(win & 15) * 1024 * 1024 + (long)rowInWin * 1024
                            + (long)(lhi * 8) * 1024;

  // Cooperative async staging: 1024 K-b128s + 1024 V-b128s, 8 per thread.
  auto stage = [&](int bb, int kb) {
#pragma unroll
    for (int c = 0; c < 4; ++c) {
      int li  = tid + c * 256;
      int key = li >> 5;                 // 0..31
      int fo  = (li & 31) << 3;          // feat offset, 8 bf16 per b128
      unsigned lk = (unsigned)(uintptr_t)&kbuf[bb][key * 256 + fo];
      const bf16* gk = kbase + (long)(kb + key) * 256 + fo;
      asm volatile("global_load_async_to_lds_b128 %0, %1, off"
                   :: "v"(lk), "v"(gk) : "memory");
      int feat = li >> 2;                // 0..255
      int ko   = (li & 3) << 3;          // key offset, 8 bf16 per b128
      unsigned lv = (unsigned)(uintptr_t)&vbuf[bb][feat * 32 + ko];
      const bf16* gv = vbase + (long)feat * 1024 + kb + ko;
      asm volatile("global_load_async_to_lds_b128 %0, %1, off"
                   :: "v"(lv), "v"(gv) : "memory");
    }
  };

  // Preload the wave's Q fragments (16 rows x 256 feats = 8 A-frags).
  v16bf qa[8];
#pragma unroll
  for (int f = 0; f < 8; ++f) {
    const bf16* p = qbase + f * 32 + lhi * 8;
    v8bf lo = *(const v8bf*)p;
    v8bf hi = *(const v8bf*)(p + 16);
#pragma unroll
    for (int e = 0; e < 8; ++e) { qa[f][e] = lo[e]; qa[f][e + 8] = hi[e]; }
  }

  const v8f zero8 = {0.f, 0.f, 0.f, 0.f, 0.f, 0.f, 0.f, 0.f};
  v8f o[16];
#pragma unroll
  for (int n = 0; n < 16; ++n) o[n] = zero8;
  float rmax[8], rsum[8];
#pragma unroll
  for (int q = 0; q < 8; ++q) { rmax[q] = -1e30f; rsum[q] = 0.0f; }

  stage(0, 0);
  for (int kb = 0; kb < 1024; kb += 32) {
    const int bb = (kb >> 5) & 1;
    asm volatile("s_wait_asynccnt 0x0" ::: "memory");  // this wave's DMAs landed
    __syncthreads();                                   // all waves' DMAs landed
    if (kb + 32 < 1024) stage(bb ^ 1, kb + 32);        // overlap next tile DMA

    // ---- S = Q @ K^T: per half, batch-load 8 K B-frags then 8-WMMA burst
    v8f s[2];
    s[0] = zero8; s[1] = zero8;
#pragma unroll
    for (int half = 0; half < 2; ++half) {
      v16bf kf[8];
#pragma unroll
      for (int f = 0; f < 8; ++f) {
        const bf16* p = &kbuf[bb][(half * 16 + lrow) * 256 + f * 32 + lhi * 8];
        v8bf lo = *(const v8bf*)p;
        v8bf hi = *(const v8bf*)(p + 16);
#pragma unroll
        for (int e = 0; e < 8; ++e) { kf[f][e] = lo[e]; kf[f][e + 8] = hi[e]; }
      }
#pragma unroll
      for (int f = 0; f < 8; ++f)
        s[half] = __builtin_amdgcn_wmma_f32_16x16x32_bf16(
            false, qa[f], false, kf[f], (short)0, s[half], false, false);
    }
    // ---- add shifted-window mask
#pragma unroll
    for (int half = 0; half < 2; ++half) {
      const float* mp = mbase + kb + half * 16 + lrow;
#pragma unroll
      for (int q = 0; q < 8; ++q) s[half][q] += mp[(long)q * 1024];
    }
    // ---- online softmax: row max / correction / exp / row sum
    float nmax[8], corr[8];
#pragma unroll
    for (int q = 0; q < 8; ++q) {
      float v = fmaxf(s[0][q], s[1][q]);
#pragma unroll
      for (int off = 1; off < 16; off <<= 1) v = fmaxf(v, __shfl_xor(v, off, 32));
      nmax[q] = fmaxf(rmax[q], v);
      corr[q] = __expf(rmax[q] - nmax[q]);
      rmax[q] = nmax[q];
    }
#pragma unroll
    for (int q = 0; q < 8; ++q) {
      float p0 = __expf(s[0][q] - nmax[q]);
      float p1 = __expf(s[1][q] - nmax[q]);
      s[0][q] = p0; s[1][q] = p1;
      float rs = p0 + p1;
#pragma unroll
      for (int off = 1; off < 16; off <<= 1) rs += __shfl_xor(rs, off, 32);
      rsum[q] = rsum[q] * corr[q] + rs;
    }
#pragma unroll
    for (int n = 0; n < 16; ++n)
#pragma unroll
      for (int q = 0; q < 8; ++q) o[n][q] *= corr[q];

    // ---- P: C-layout (f32 frags) -> A-layout bf16 via wave-private LDS
    bf16* pl = &plds[wave][0];
#pragma unroll
    for (int half = 0; half < 2; ++half)
#pragma unroll
      for (int q = 0; q < 8; ++q)
        pl[(lhi * 8 + q) * 32 + half * 16 + lrow] = (bf16)s[half][q];
    v16bf pa;
    {
      v8bf lo = *(const v8bf*)(pl + lrow * 32 + lhi * 8);
      v8bf hi = *(const v8bf*)(pl + lrow * 32 + 16 + lhi * 8);
#pragma unroll
      for (int e = 0; e < 8; ++e) { pa[e] = lo[e]; pa[e + 8] = hi[e]; }
    }
    // ---- O += P @ V: batch-load 4 V B-frags per group, then 4-WMMA burst
#pragma unroll
    for (int g = 0; g < 4; ++g) {
      v16bf vf[4];
#pragma unroll
      for (int t = 0; t < 4; ++t) {
        const bf16* p = &vbuf[bb][((g * 4 + t) * 16 + lrow) * 32 + lhi * 8];
        v8bf lo = *(const v8bf*)p;
        v8bf hi = *(const v8bf*)(p + 16);
#pragma unroll
        for (int e = 0; e < 8; ++e) { vf[t][e] = lo[e]; vf[t][e + 8] = hi[e]; }
      }
#pragma unroll
      for (int t = 0; t < 4; ++t)
        o[g * 4 + t] = __builtin_amdgcn_wmma_f32_16x16x32_bf16(
            false, pa, false, vf[t], (short)0, o[g * 4 + t], false, false);
    }
    __syncthreads();  // all waves done reading buffer bb before it is reused
  }

  // ---- epilogue: normalize and store (window-linear rows, bf16)
#pragma unroll
  for (int q = 0; q < 8; ++q) rsum[q] = (rsum[q] > 0.f) ? 1.0f / rsum[q] : 0.f;
  const long r0 = (long)(win * 1024 + rowInWin);
#pragma unroll
  for (int n = 0; n < 16; ++n)
#pragma unroll
    for (int q = 0; q < 8; ++q)
      out[(r0 + lhi * 8 + q) * 256 + n * 16 + lrow] = (bf16)(o[n][q] * rsum[q]);
}

// ---------------------------------------------------------------------------
// LayerNorm over 256 cols, one wave per row (8 rows per block).
//   outbf != null : write bf16 into hcat right half  (LN1)
//   outf  != null : write f32 resid + LN             (LN2 + residual)
// ---------------------------------------------------------------------------
__launch_bounds__(256)
__global__ void layernorm_k(const float* __restrict__ x, const float* __restrict__ gamma,
                            const float* __restrict__ beta,
                            bf16* __restrict__ outbf, int ldo, int off,
                            const float* __restrict__ resid, float* __restrict__ outf) {
  int row  = blockIdx.x * 8 + (threadIdx.x >> 5);
  int lane = threadIdx.x & 31;
  const float* xr = x + (long)row * 256;
  float v[8], s = 0.f, ss = 0.f;
#pragma unroll
  for (int t = 0; t < 8; ++t) {
    v[t] = xr[lane + 32 * t];
    s += v[t]; ss += v[t] * v[t];
  }
#pragma unroll
  for (int o2 = 1; o2 < 32; o2 <<= 1) {
    s  += __shfl_xor(s, o2, 32);
    ss += __shfl_xor(ss, o2, 32);
  }
  float mean = s * (1.0f / 256.0f);
  float var  = ss * (1.0f / 256.0f) - mean * mean;
  float rstd = rsqrtf(var + 1e-5f);
#pragma unroll
  for (int t = 0; t < 8; ++t) {
    int c = lane + 32 * t;
    float y = (v[t] - mean) * rstd * gamma[c] + beta[c];
    if (outbf) outbf[(long)row * ldo + off + c] = (bf16)y;
    if (outf)  outf[(long)row * 256 + c] = resid[(long)row * 256 + c] + y;
  }
}

// ---------------------------------------------------------------------------
// Host-side orchestration
// ---------------------------------------------------------------------------
extern "C" void kernel_launch(void* const* d_in, const int* in_sizes, int n_in,
                              void* d_out, int out_size, void* d_ws, size_t ws_size,
                              hipStream_t stream) {
  const float* source = (const float*)d_in[0];
  const float* target = (const float*)d_in[1];
  const float* mask   = (const float*)d_in[2];
  const float* Wq     = (const float*)d_in[3];
  const float* Wk     = (const float*)d_in[4];
  const float* Wv     = (const float*)d_in[5];
  const float* Wm     = (const float*)d_in[6];
  const float* g1     = (const float*)d_in[7];
  const float* b1     = (const float*)d_in[8];
  const float* W1     = (const float*)d_in[9];
  const float* W2     = (const float*)d_in[10];
  const float* g2     = (const float*)d_in[11];
  const float* b2     = (const float*)d_in[12];
  float* out = (float*)d_out;

  size_t off = 0;
  auto alloc = [&](size_t bytes) {
    size_t p = off;
    off = (off + bytes + 255) & ~(size_t)255;
    return p;
  };
  char* ws = (char*)d_ws;
  bf16* hcat = (bf16*)(ws + alloc((size_t)MROWS * DCAT * 2));   // [32768,512] bf16
  bf16* Xt   = (bf16*)(ws + alloc((size_t)MROWS * DIM * 2));
  bf16* Wqt  = (bf16*)(ws + alloc((size_t)DIM * DIM * 2));
  bf16* Wkt  = (bf16*)(ws + alloc((size_t)DIM * DIM * 2));
  bf16* Wvt  = (bf16*)(ws + alloc((size_t)DIM * DIM * 2));
  bf16* Wmt  = (bf16*)(ws + alloc((size_t)DIM * DIM * 2));
  bf16* W1t  = (bf16*)(ws + alloc((size_t)DHID * DCAT * 2));    // [2048,512]
  bf16* W2t  = (bf16*)(ws + alloc((size_t)DIM * DHID * 2));     // [256,2048]
  bf16* qw   = (bf16*)(ws + alloc((size_t)MROWS * DIM * 2));    // window-linear
  bf16* kw   = (bf16*)(ws + alloc((size_t)MROWS * DIM * 2));
  bf16* vt   = (bf16*)(ws + alloc((size_t)MROWS * DIM * 2));    // [win][feat][key]
  bf16* attn = (bf16*)(ws + alloc((size_t)MROWS * DIM * 2));
  float* msg = (float*)(ws + alloc((size_t)MROWS * DIM * 4));   // reused for fc2 out
  bf16* h1   = (bf16*)(ws + alloc((size_t)MROWS * DHID * 2));   // gelu(fc1)

  const dim3 blk(256);
  // 1) bf16 conversions (source -> hcat left half, stride 512)
  cvt_strided<<<dim3((MROWS * DIM) / 256), blk, 0, stream>>>(source, hcat, MROWS, DIM, DCAT, 0);
  cvt_strided<<<dim3((MROWS * DIM) / 256), blk, 0, stream>>>(target, Xt, MROWS, DIM, DIM, 0);
  cvt_transpose<<<dim3((DIM * DIM) / 256), blk, 0, stream>>>(Wq, Wqt, DIM, DIM);
  cvt_transpose<<<dim3((DIM * DIM) / 256), blk, 0, stream>>>(Wk, Wkt, DIM, DIM);
  cvt_transpose<<<dim3((DIM * DIM) / 256), blk, 0, stream>>>(Wv, Wvt, DIM, DIM);
  cvt_transpose<<<dim3((DIM * DIM) / 256), blk, 0, stream>>>(Wm, Wmt, DIM, DIM);
  cvt_transpose<<<dim3((DCAT * DHID) / 256), blk, 0, stream>>>(W1, W1t, DCAT, DHID);
  cvt_transpose<<<dim3((DHID * DIM) / 256), blk, 0, stream>>>(W2, W2t, DHID, DIM);

  // 2) QKV projections with fused shift+window gather; q pre-scaled by 1/sqrt(d)
  gemm_wmma<1, 0, 0><<<dim3(MROWS / 128, DIM / 128), blk, 0, stream>>>(
      hcat, DCAT, Wqt, DIM, qw, DIM, 0.0625f);
  gemm_wmma<1, 0, 0><<<dim3(MROWS / 128, DIM / 128), blk, 0, stream>>>(
      Xt, DIM, Wkt, DIM, kw, DIM, 1.0f);
  gemm_wmma<1, 3, 0><<<dim3(MROWS / 128, DIM / 128), blk, 0, stream>>>(
      Xt, DIM, Wvt, DIM, vt, DIM, 1.0f);   // store V^T per window

  // 3) windowed flash attention (32 windows x 8 row-slabs)
  swin_attn<<<dim3(NWIN * 8), blk, 0, stream>>>(qw, kw, vt, mask, attn);

  // 4) merge projection + fused un-window/un-shift scatter (f32)
  gemm_wmma<0, 2, 0><<<dim3(MROWS / 128, DIM / 128), blk, 0, stream>>>(
      attn, DIM, Wmt, DIM, msg, DIM, 1.0f);

  // 5) LN1 -> bf16 into hcat right half
  layernorm_k<<<dim3(MROWS / 8), blk, 0, stream>>>(msg, g1, b1, hcat, DCAT, DIM,
                                                   nullptr, nullptr);

  // 6) MLP: gelu(hcat @ W1) -> h1 (bf16), then h1 @ W2 -> msg (f32)
  gemm_wmma<0, 0, 1><<<dim3(MROWS / 128, DHID / 128), blk, 0, stream>>>(
      hcat, DCAT, W1t, DCAT, h1, DHID, 1.0f);
  gemm_wmma<0, 1, 0><<<dim3(MROWS / 128, DIM / 128), blk, 0, stream>>>(
      h1, DHID, W2t, DHID, msg, DIM, 1.0f);

  // 7) LN2 + residual -> f32 output
  layernorm_k<<<dim3(MROWS / 8), blk, 0, stream>>>(msg, g2, b2, nullptr, 0, 0,
                                                   source, out);
  (void)in_sizes; (void)n_in; (void)out_size; (void)ws_size;
}
